// MSDGNN_block_71408126263416
// MI455X (gfx1250) — compile-verified, matrix operand order checked by reading, not run on hardware
//
#include <hip/hip_runtime.h>
#include <hip/hip_bf16.h>
#include <math.h>

// ---------------------------------------------------------------------------
// ASTGCN block, fully fused: one workgroup per batch element.
//   temporal attention -> x_TAt -> spatial attention -> cheb graph conv (WMMA)
//   -> 1x3 temporal conv + 1x1 residual conv (WMMA, t-major im2col in LDS)
//   -> ReLU -> LayerNorm(ch) -> output [B,N,TIMEF,T]
// Single pass over x (92MB read) + output (184MB write): HBM roofline ~12us.
// All GEMM-shaped math on v_wmma_f32_16x16x32_f16; all WMMA operands are
// k-contiguous in LDS (ds_load_b128 merges). Tile map is division-free:
// each wave owns one column tile (ct = wave&3), so each B fragment is loaded
// once per K-chunk and reused across the wave's 11 row tiles.
// ---------------------------------------------------------------------------

typedef _Float16 half16 __attribute__((ext_vector_type(16)));
typedef _Float16 half4v __attribute__((ext_vector_type(4)));
typedef float    float8 __attribute__((ext_vector_type(8)));
typedef float    float4v __attribute__((ext_vector_type(4)));

#define NB     22
#define FIN    32
#define TDIM   32
#define NCHEV  64
#define NTIMEF 64
#define XTILE  (NB * FIN * TDIM)   /* 22528 x elements per batch      */
#define NTHREADS 512
#define ROWS   (NB * TDIM)         /* 704 output rows (n,t)           */
#define RT_TILES 44                /* 704/16                          */
#define TPW    11                  /* row tiles per wave (44/4 groups)*/
#define LN_EPS 1e-5f

#define XH_S   36                  /* XH [n][f][36] half (8B-aligned t-quads) */
#define GPT_T  68                  /* gpt t-stride in halfs (conflict-free)   */
#define GPT_N  (34 * GPT_T)        /* gpt n-stride (34 padded t rows)         */

// ------------------------- LDS layout (byte offsets) ------------------------
static constexpr int OFF_SUP1  = 0;                    // 484 f32
static constexpr int OFF_SUP2  = 1936;                 // 484 f32
static constexpr int OFF_SNORM = 3872;                 // 484 f32
static constexpr int OFF_AK    = 5808;                 // 484 f32
static constexpr int OFF_BIAS  = 7744;                 // 64 f32 (tb+rb)
static constexpr int OFF_LNG   = 8000;                 // 64 f32
static constexpr int OFF_LNB   = 8256;                 // 64 f32
static constexpr int OFF_WTT   = 8512;                 // [64][192] f16 (o,dt,c)
static constexpr int OFF_WRT   = 33088;                // [64][32]  f16 (o,f)
static constexpr int OFF_THT   = 37184;                // [64][96]  f16 (o,k*32+f)
static constexpr int OFF_XH    = 49472;                // [22][32][36] f16 (f-major)
static constexpr int OFF_XT    = 100160;               // [22][32][32] f16 (t-major)
static constexpr int OFF_RHSB  = 145216;               // [704][33] f16 (cheb rhs)
static constexpr int OFF_GPT   = 191680;               // [22][34][68] f16 (t-major gcn, T halo)
static constexpr int SMEM_TOTAL = 293408;

// attention scratch aliases RHSB.. (dead before cheb stage)
static constexpr int A_PU1   = OFF_RHSB + 0;
static constexpr int A_PU2   = OFF_RHSB + 96;      // [32][22] f32
static constexpr int A_PU3   = OFF_RHSB + 2912;
static constexpr int A_PBE   = OFF_RHSB + 3040;    // [32][32]
static constexpr int A_PVE   = OFF_RHSB + 7136;    // [32][32]
static constexpr int A_PW1   = OFF_RHSB + 11232;
static constexpr int A_PW2   = OFF_RHSB + 11360;   // [32][32]
static constexpr int A_PW3   = OFF_RHSB + 15456;
static constexpr int A_PVS   = OFF_RHSB + 15584;   // [22][22]
static constexpr int A_PBS   = OFF_RHSB + 17520;   // [22][22]
static constexpr int A_LHST  = OFF_RHSB + 19456;   // [32][32]
static constexpr int A_TMPTN = OFF_RHSB + 23552;   // [32][24]
static constexpr int A_RHST  = OFF_RHSB + 26624;   // [22][33]
static constexpr int A_PRODE = OFF_RHSB + 29536;   // [32][33]
static constexpr int A_ESIG  = OFF_RHSB + 33760;   // [32][33]
static constexpr int A_EN    = OFF_RHSB + 37984;   // [32][33]
static constexpr int A_LHSSA = OFF_RHSB + 42208;   // [22][32]
static constexpr int A_RHSS  = OFF_RHSB + 45024;   // [22][33]
static constexpr int A_LHS2  = OFF_RHSB + 47936;   // [22][33]
static constexpr int A_PRODS = OFF_RHSB + 50848;   // [22][22]
static constexpr int A_SSIG  = OFF_RHSB + 52784;   // [22][22]
static constexpr int A_GT    = OFF_RHSB + 54720;   // [22][22]
// LayerNorm staging (after all WMMA done; aliases rhsb+gpt head)
static constexpr int OFF_YOUT = OFF_RHSB;          // [704][66] f16

// ---------------------------------------------------------------------------
// Fragment loader per CDNA5 ISA 7.12.2 (16-bit A 16x32 layout; B fragments are
// loaded from a transposed [N][K] row-major view with the symmetric layout).
// element(m, k) = base[m*mstride + k]   (k always contiguous -> b64/b128 merge)
__device__ __forceinline__ half16 ld_frag(const _Float16* base, int mstride) {
  const int lane = threadIdx.x & 31;
  const int mr   = lane & 15;
  const int hf   = lane >> 4;
  half16 r;
#pragma unroll
  for (int v = 0; v < 8; ++v) {
    const int kb = ((v & 4) ? 16 : 0) + hf * 8 + (v & 3) * 2;
    const _Float16* p = base + mr * mstride + kb;
    r[2 * v]     = p[0];
    r[2 * v + 1] = p[1];
  }
  return r;
}

__device__ __forceinline__ float8 wmma32(half16 a, half16 b, float8 c) {
  return __builtin_amdgcn_wmma_f32_16x16x32_f16(false, a, false, b, (short)0, c,
                                                false, false);
}

__device__ __forceinline__ float sigmoidf_(float z) { return 1.0f / (1.0f + __expf(-z)); }

// ---------------------------------------------------------------------------
__global__ __launch_bounds__(NTHREADS)
void astgcn_fused_kernel(const float* __restrict__ x,
                         const float* __restrict__ U1, const float* __restrict__ U2,
                         const float* __restrict__ U3, const float* __restrict__ be,
                         const float* __restrict__ Ve, const float* __restrict__ W1,
                         const float* __restrict__ W2, const float* __restrict__ W3,
                         const float* __restrict__ bs, const float* __restrict__ Vs,
                         const float* __restrict__ Theta, const float* __restrict__ node_emb,
                         const float* __restrict__ tconv_w, const float* __restrict__ tconv_b,
                         const float* __restrict__ rconv_w, const float* __restrict__ rconv_b,
                         const float* __restrict__ ln_g, const float* __restrict__ ln_b,
                         float* __restrict__ out) {
  extern __shared__ char smem[];
  const int tid  = threadIdx.x;
  const int wave = tid >> 5;
  const int b    = blockIdx.x;
  // division-free tile map: fixed column tile per wave, strided row tiles
  const int ctw  = wave & 3;         // column tile (o-chunk) of this wave
  const int w4   = wave >> 2;        // row-tile group

  float*     SUP1f  = (float*)(smem + OFF_SUP1);
  float*     SUP2f  = (float*)(smem + OFF_SUP2);
  float*     SNORMf = (float*)(smem + OFF_SNORM);
  float*     AKf    = (float*)(smem + OFF_AK);
  float*     BIASf  = (float*)(smem + OFF_BIAS);
  float*     LNGf   = (float*)(smem + OFF_LNG);
  float*     LNBf   = (float*)(smem + OFF_LNB);
  _Float16*  WTT    = (_Float16*)(smem + OFF_WTT);
  _Float16*  WRT    = (_Float16*)(smem + OFF_WRT);
  _Float16*  THT    = (_Float16*)(smem + OFF_THT);
  _Float16*  XH     = (_Float16*)(smem + OFF_XH);
  _Float16*  XT     = (_Float16*)(smem + OFF_XT);
  _Float16*  RHSB   = (_Float16*)(smem + OFF_RHSB);
  _Float16*  GPT    = (_Float16*)(smem + OFF_GPT);
  _Float16*  YOUT   = (_Float16*)(smem + OFF_YOUT);

  // ---------------- Phase 0: stage x tile + weights + attn params ----------
  if (tid == 0) {
    __builtin_prefetch(x + (size_t)b * XTILE, 0, 0);
    __builtin_prefetch(tconv_w, 0, 0);
  }
  const float* xb = x + (size_t)b * XTILE;
  for (int e2 = tid; e2 < XTILE / 4; e2 += NTHREADS) { // x[b] -> XH (f-major) + XT (t-major)
    const int e = e2 << 2;
    const int n = e >> 10, rem = e & 1023, f = rem >> 5, t0 = rem & 31;
    const float4v xv = *(const float4v*)(xb + e);      // global_load_b128
    half4v hv;
#pragma unroll
    for (int j = 0; j < 4; ++j) hv[j] = (_Float16)xv[j];
    *(half4v*)(XH + (n * FIN + f) * XH_S + t0) = hv;   // 8B-aligned ds_store_b64
#pragma unroll
    for (int j = 0; j < 4; ++j) XT[(n * TDIM + t0 + j) * 32 + f] = hv[j];
  }
  for (int e = tid; e < NTIMEF * 192; e += NTHREADS) { // tconv w -> [o][dt][c]
    const int o = e / 192, rem = e % 192, dt = rem / 64, c = rem % 64;
    WTT[e] = (_Float16)tconv_w[(o * NCHEV + c) * 3 + dt];
  }
  for (int e = tid; e < NTIMEF * FIN; e += NTHREADS)   // rconv w already [o][f]
    WRT[e] = (_Float16)rconv_w[e];
  for (int e = tid; e < NTIMEF * 96; e += NTHREADS) {  // Theta -> [o][k*32+f]
    const int o = e / 96, kf = e % 96;
    THT[e] = (_Float16)Theta[kf * NTIMEF + o];
  }
  for (int e = tid; e < NTIMEF; e += NTHREADS) {
    BIASf[e] = tconv_b[e] + rconv_b[e];
    LNGf[e]  = ln_g[e];
    LNBf[e]  = ln_b[e];
  }
  { // attention params into LDS
    float* p;
    p = (float*)(smem + A_PU1); for (int e = tid; e < NB;     e += NTHREADS) p[e] = U1[e];
    p = (float*)(smem + A_PU2); for (int e = tid; e < FIN*NB; e += NTHREADS) p[e] = U2[e];
    p = (float*)(smem + A_PU3); for (int e = tid; e < FIN;    e += NTHREADS) p[e] = U3[e];
    p = (float*)(smem + A_PBE); for (int e = tid; e < 1024;   e += NTHREADS) p[e] = be[e];
    p = (float*)(smem + A_PVE); for (int e = tid; e < 1024;   e += NTHREADS) p[e] = Ve[e];
    p = (float*)(smem + A_PW1); for (int e = tid; e < TDIM;   e += NTHREADS) p[e] = W1[e];
    p = (float*)(smem + A_PW2); for (int e = tid; e < 1024;   e += NTHREADS) p[e] = W2[e];
    p = (float*)(smem + A_PW3); for (int e = tid; e < FIN;    e += NTHREADS) p[e] = W3[e];
    p = (float*)(smem + A_PVS); for (int e = tid; e < 484;    e += NTHREADS) p[e] = Vs[e];
    p = (float*)(smem + A_PBS); for (int e = tid; e < 484;    e += NTHREADS) p[e] = bs[e];
    p = (float*)(smem + A_RHSS); for (int e = tid; e < NB * 33; e += NTHREADS) p[e] = 0.f;
  }
  __syncthreads();

  // ---------------- Phase 0b: adaptive supports (recomputed per block) -----
  {
    float* GT = (float*)(smem + A_GT);
    for (int e = tid; e < 484; e += NTHREADS) {        // relu(ne @ ne^T)
      const int n = e / NB, m = e % NB;
      float a = 0.f;
#pragma unroll
      for (int p2 = 0; p2 < NB; ++p2) a += node_emb[n * NB + p2] * node_emb[m * NB + p2];
      GT[e] = fmaxf(a, 0.f);
    }
    __syncthreads();
    if (tid < NB) {                                    // row softmax (axis=1)
      float mx = -1e30f;
      for (int m = 0; m < NB; ++m) mx = fmaxf(mx, GT[tid * NB + m]);
      float s = 0.f;
      for (int m = 0; m < NB; ++m) s += __expf(GT[tid * NB + m] - mx);
      const float inv = 1.f / s;
      for (int m = 0; m < NB; ++m) SUP1f[tid * NB + m] = __expf(GT[tid * NB + m] - mx) * inv;
    }
    __syncthreads();
    for (int e = tid; e < 484; e += NTHREADS) {        // T2 = 2*S@S - I
      const int n = e / NB, m = e % NB;
      float a = 0.f;
#pragma unroll
      for (int p2 = 0; p2 < NB; ++p2) a += SUP1f[n * NB + p2] * SUP1f[p2 * NB + m];
      SUP2f[e] = 2.f * a - ((n == m) ? 1.f : 0.f);
    }
    __syncthreads();
  }

  // ---------------- Phase 1: temporal attention -> E_norm ------------------
  {
    float* PU1 = (float*)(smem + A_PU1);  float* PU2 = (float*)(smem + A_PU2);
    float* PU3 = (float*)(smem + A_PU3);  float* PBE = (float*)(smem + A_PBE);
    float* PVE = (float*)(smem + A_PVE);
    float* LHST = (float*)(smem + A_LHST);   float* TMPTN = (float*)(smem + A_TMPTN);
    float* RHST = (float*)(smem + A_RHST);   float* PRODE = (float*)(smem + A_PRODE);
    float* ESIG = (float*)(smem + A_ESIG);   float* EN    = (float*)(smem + A_EN);

    for (int e = tid; e < 1024; e += NTHREADS) {       // lhs_t[t][f]
      const int t = e >> 5, f = e & 31;
      float a = 0.f;
#pragma unroll
      for (int n = 0; n < NB; ++n) a += (float)XH[(n * FIN + f) * XH_S + t] * PU1[n];
      LHST[t * 32 + f] = a;
    }
    __syncthreads();
    for (int e = tid; e < TDIM * NB; e += NTHREADS) {  // tmp[t][n] = lhs_t @ U2
      const int t = e / NB, n = e % NB;
      float a = 0.f;
#pragma unroll
      for (int f = 0; f < FIN; ++f) a += LHST[t * 32 + f] * PU2[f * NB + n];
      TMPTN[t * 24 + n] = a;
    }
    for (int e = tid; e < NB * TDIM; e += NTHREADS) {  // rhs_t[n][t]
      const int n = e >> 5, t = e & 31;
      float a = 0.f;
#pragma unroll
      for (int f = 0; f < FIN; ++f) a += PU3[f] * (float)XH[(n * FIN + f) * XH_S + t];
      RHST[n * 33 + t] = a;
    }
    __syncthreads();
    for (int e = tid; e < 1024; e += NTHREADS) {       // prod[t][s]
      const int t = e >> 5, s = e & 31;
      float a = 0.f;
#pragma unroll
      for (int n = 0; n < NB; ++n) a += TMPTN[t * 24 + n] * RHST[n * 33 + s];
      PRODE[t * 33 + s] = a;
    }
    __syncthreads();
    for (int e = tid; e < 1024; e += NTHREADS) {       // sigmoid(prod + be)
      const int s = e >> 5, u = e & 31;
      ESIG[s * 33 + u] = sigmoidf_(PRODE[s * 33 + u] + PBE[s * 32 + u]);
    }
    __syncthreads();
    for (int e = tid; e < 1024; e += NTHREADS) {       // E = Ve @ sig
      const int t = e >> 5, u = e & 31;
      float a = 0.f;
#pragma unroll
      for (int s = 0; s < TDIM; ++s) a += PVE[t * 32 + s] * ESIG[s * 33 + u];
      PRODE[t * 33 + u] = a;
    }
    __syncthreads();
    if (tid < TDIM) {                                  // softmax over dim t (axis=1)
      const int u = tid;
      float mx = -1e30f;
      for (int t = 0; t < TDIM; ++t) mx = fmaxf(mx, PRODE[t * 33 + u]);
      float s = 0.f;
      for (int t = 0; t < TDIM; ++t) s += __expf(PRODE[t * 33 + u] - mx);
      const float inv = 1.f / s;
      for (int t = 0; t < TDIM; ++t) EN[t * 33 + u] = __expf(PRODE[t * 33 + u] - mx) * inv;
    }
    __syncthreads();
  }

  // ---------------- Phase 2: x_TAt (implicit) -> spatial attention ---------
  {
    float* PW1 = (float*)(smem + A_PW1);  float* PW2 = (float*)(smem + A_PW2);
    float* PW3 = (float*)(smem + A_PW3);  float* PVS = (float*)(smem + A_PVS);
    float* PBS = (float*)(smem + A_PBS);  float* EN  = (float*)(smem + A_EN);
    float* LHSSA = (float*)(smem + A_LHSSA); float* RHSS = (float*)(smem + A_RHSS);
    float* LHS2  = (float*)(smem + A_LHS2);  float* PRODS = (float*)(smem + A_PRODS);
    float* SSIG  = (float*)(smem + A_SSIG);  float* SBUF  = (float*)(smem + A_GT);

    for (int r = tid; r < NB * FIN; r += NTHREADS) {   // row (n,f) of x_TAt
      const int n = r >> 5, f = r & 31;
      float xr[TDIM];
#pragma unroll
      for (int s = 0; s < TDIM; ++s) xr[s] = (float)XH[(n * FIN + f) * XH_S + s];
      const float w3f = PW3[f];
      float lh = 0.f;
      for (int t = 0; t < TDIM; ++t) {
        float v = 0.f;
#pragma unroll
        for (int s = 0; s < TDIM; ++s) v += xr[s] * EN[s * 33 + t];
        lh += v * PW1[t];
        atomicAdd(&RHSS[n * 33 + t], w3f * v);         // ds_add_f32
      }
      LHSSA[n * 32 + f] = lh;
    }
    __syncthreads();
    for (int e = tid; e < NB * TDIM; e += NTHREADS) {  // lhs2 = lhs_s @ W2
      const int n = e >> 5, t = e & 31;
      float a = 0.f;
#pragma unroll
      for (int f = 0; f < FIN; ++f) a += LHSSA[n * 32 + f] * PW2[f * 32 + t];
      LHS2[n * 33 + t] = a;
    }
    __syncthreads();
    for (int e = tid; e < 484; e += NTHREADS) {        // prod_s[n][m]
      const int n = e / NB, m = e % NB;
      float a = 0.f;
#pragma unroll
      for (int t = 0; t < TDIM; ++t) a += LHS2[n * 33 + t] * RHSS[m * 33 + t];
      PRODS[n * NB + m] = a;
    }
    __syncthreads();
    for (int e = tid; e < 484; e += NTHREADS)
      SSIG[e] = sigmoidf_(PRODS[e] + PBS[e]);
    __syncthreads();
    for (int e = tid; e < 484; e += NTHREADS) {        // S = Vs @ sig
      const int n = e / NB, l = e % NB;
      float a = 0.f;
#pragma unroll
      for (int m = 0; m < NB; ++m) a += PVS[n * NB + m] * SSIG[m * NB + l];
      SBUF[n * NB + l] = a;
    }
    __syncthreads();
    if (tid < NB) {                                    // softmax over n (axis=1)
      const int l = tid;
      float mx = -1e30f;
      for (int n = 0; n < NB; ++n) mx = fmaxf(mx, SBUF[n * NB + l]);
      float s = 0.f;
      for (int n = 0; n < NB; ++n) s += __expf(SBUF[n * NB + l] - mx);
      const float inv = 1.f / s;
      for (int n = 0; n < NB; ++n) SNORMf[n * NB + l] = __expf(SBUF[n * NB + l] - mx) * inv;
    }
    __syncthreads();
  }

  // attention scratch is dead: zero gpt T-halo (rows tp=0 and tp=33)
  for (int e = tid; e < NB * NCHEV; e += NTHREADS) {
    const int n = e >> 6, c = e & 63;
    GPT[n * GPT_N + c]              = (_Float16)0.f;
    GPT[n * GPT_N + 33 * GPT_T + c] = (_Float16)0.f;
  }

  // ---------------- Phase 3: cheb conv (WMMA over Theta) -------------------
  float8 acc[TPW];
#pragma unroll
  for (int i = 0; i < TPW; ++i)
#pragma unroll
    for (int v = 0; v < 8; ++v) acc[i][v] = 0.f;

#pragma unroll
  for (int k = 0; k < 3; ++k) {
    __syncthreads();
    for (int e = tid; e < 484; e += NTHREADS) {        // A_k = sup_k (*) S_norm
      const int n = e / NB, m = e % NB;
      const float sk = (k == 0) ? ((n == m) ? 1.f : 0.f) : (k == 1 ? SUP1f[e] : SUP2f[e]);
      AKf[e] = sk * SNORMf[e];
    }
    __syncthreads();
    for (int e2 = tid; e2 < XTILE / 4; e2 += NTHREADS) { // rhs_k[(m,t)][f] f16, 4 t/iter
      const int m = e2 >> 8, rem = e2 & 255, f = rem >> 3, t0 = (rem & 7) << 2;
      float a0 = 0.f, a1 = 0.f, a2 = 0.f, a3 = 0.f;
#pragma unroll
      for (int n = 0; n < NB; ++n) {
        const float ak = AKf[n * NB + m];
        const half4v hv = *(const half4v*)(XH + (n * FIN + f) * XH_S + t0); // ds_load_b64
        a0 += ak * (float)hv[0];
        a1 += ak * (float)hv[1];
        a2 += ak * (float)hv[2];
        a3 += ak * (float)hv[3];
      }
      RHSB[(m * TDIM + t0 + 0) * 33 + f] = (_Float16)a0;
      RHSB[(m * TDIM + t0 + 1) * 33 + f] = (_Float16)a1;
      RHSB[(m * TDIM + t0 + 2) * 33 + f] = (_Float16)a2;
      RHSB[(m * TDIM + t0 + 3) * 33 + f] = (_Float16)a3;
    }
    __syncthreads();
    {                                                  // one B fragment per k
      const half16 bm = ld_frag(THT + ctw * 16 * 96 + k * 32, 96);
#pragma unroll
      for (int i = 0; i < TPW; ++i) {                  // C += rhs_k @ Theta_k
        const int rt = w4 + 4 * i;
        const half16 a = ld_frag(RHSB + rt * 16 * 33, 33);
        acc[i] = wmma32(a, bm, acc[i]);
      }
    }
  }
  __syncthreads();
  {                                                    // relu -> gpt[n][1+t][c]
    const int lane = tid & 31, hf = lane >> 4;
    const int col = ctw * 16 + (lane & 15);
#pragma unroll
    for (int i = 0; i < TPW; ++i) {
      const int rt = w4 + 4 * i;
      const int n0 = rt >> 1, t0 = (rt & 1) * 16;
#pragma unroll
      for (int v = 0; v < 8; ++v) {
        const int tt = t0 + v + 8 * hf;
        GPT[n0 * GPT_N + (1 + tt) * GPT_T + col] = (_Float16)fmaxf(acc[i][v], 0.f);
      }
    }
  }
  __syncthreads();

  // ---------------- Phase 4: 1x3 tconv + 1x1 rconv (WMMA) ------------------
  {
    const int lane = tid & 31;
    const int col  = ctw * 16 + (lane & 15);
    const float bv = BIASf[col];
#pragma unroll
    for (int i = 0; i < TPW; ++i)                      // C init = biases
#pragma unroll
      for (int v = 0; v < 8; ++v) acc[i][v] = bv;
#pragma unroll
    for (int dt = 0; dt < 3; ++dt) {
#pragma unroll
      for (int ci = 0; ci < 2; ++ci) {                 // one B fragment per chunk
        const half16 bm = ld_frag(WTT + ctw * 16 * 192 + dt * 64 + ci * 32, 192);
#pragma unroll
        for (int i = 0; i < TPW; ++i) {                // im2col view of gpt
          const int rt = w4 + 4 * i;
          const int n0 = rt >> 1, t0 = (rt & 1) * 16;
          const half16 a = ld_frag(GPT + n0 * GPT_N + (t0 + dt) * GPT_T + ci * 32, GPT_T);
          acc[i] = wmma32(a, bm, acc[i]);
        }
      }
    }
    {                                                  // residual 1x1 from XT
      const half16 bm = ld_frag(WRT + ctw * 16 * 32, 32);
#pragma unroll
      for (int i = 0; i < TPW; ++i) {
        const int rt = w4 + 4 * i;
        const int n0 = rt >> 1, t0 = (rt & 1) * 16;
        const half16 a = ld_frag(XT + n0 * TDIM * 32 + t0 * 32, 32);
        acc[i] = wmma32(a, bm, acc[i]);
      }
    }
  }
  __syncthreads();                                     // gpt/xt now dead
  {                                                    // relu -> YOUT staging
    const int lane = tid & 31, hf = lane >> 4;
    const int col = ctw * 16 + (lane & 15);
#pragma unroll
    for (int i = 0; i < TPW; ++i) {
      const int rt = w4 + 4 * i;
      const int r0 = rt * 16;
#pragma unroll
      for (int v = 0; v < 8; ++v) {
        const int r = r0 + v + 8 * hf;
        YOUT[r * 66 + col] = (_Float16)fmaxf(acc[i][v], 0.f);
      }
    }
  }
  __syncthreads();

  // ---------------- Phase 5: LayerNorm over channels + store ---------------
  for (int r = tid; r < ROWS; r += NTHREADS) {
    float s = 0.f, s2 = 0.f;
#pragma unroll
    for (int o = 0; o < NTIMEF; ++o) {
      const float v = (float)YOUT[r * 66 + o];
      s += v; s2 += v * v;
    }
    const float mu  = s * (1.0f / NTIMEF);
    const float var = s2 * (1.0f / NTIMEF) - mu * mu;
    const float inv = rsqrtf(var + LN_EPS);
    const int n = r >> 5, t = r & 31;
    float* op = out + (((size_t)b * NB + n) * NTIMEF) * TDIM + t;
#pragma unroll
    for (int o = 0; o < NTIMEF; ++o) {
      const float v = (float)YOUT[r * 66 + o];
      op[o * TDIM] = (v - mu) * inv * LNGf[o] + LNBf[o];
    }
  }
}

// ---------------------------------------------------------------------------
extern "C" void kernel_launch(void* const* d_in, const int* in_sizes, int n_in,
                              void* d_out, int out_size, void* d_ws, size_t ws_size,
                              hipStream_t stream) {
  (void)n_in; (void)d_ws; (void)ws_size; (void)out_size;
  const float* x        = (const float*)d_in[0];
  const float* U1       = (const float*)d_in[1];
  const float* U2       = (const float*)d_in[2];
  const float* U3       = (const float*)d_in[3];
  const float* be       = (const float*)d_in[4];
  const float* Ve       = (const float*)d_in[5];
  const float* W1       = (const float*)d_in[6];
  const float* W2       = (const float*)d_in[7];
  const float* W3       = (const float*)d_in[8];
  const float* bs       = (const float*)d_in[9];
  const float* Vs       = (const float*)d_in[10];
  const float* Theta    = (const float*)d_in[11];
  const float* node_emb = (const float*)d_in[12];
  const float* tconv_w  = (const float*)d_in[13];
  const float* tconv_b  = (const float*)d_in[14];
  const float* rconv_w  = (const float*)d_in[15];
  const float* rconv_b  = (const float*)d_in[16];
  const float* ln_g     = (const float*)d_in[17];
  const float* ln_b     = (const float*)d_in[18];
  float* out = (float*)d_out;

  const int B = in_sizes[0] / XTILE;

  hipFuncSetAttribute(reinterpret_cast<const void*>(astgcn_fused_kernel),
                      hipFuncAttributeMaxDynamicSharedMemorySize, SMEM_TOTAL);
  astgcn_fused_kernel<<<B, NTHREADS, SMEM_TOTAL, stream>>>(
      x, U1, U2, U3, be, Ve, W1, W2, W3, bs, Vs, Theta, node_emb,
      tconv_w, tconv_b, rconv_w, rconv_b, ln_g, ln_b, out);
}
